// second_order_RNN_24489903522725
// MI455X (gfx1250) — compile-verified, matrix operand order checked by reading, not run on hardware
//
#include <hip/hip_runtime.h>
#include <stdint.h>

// Problem constants
#define BATCH 1024
#define DIM   256
#define RANK  256
#define OUTD  128
#define LEN   64
#define KTOT  (RANK * DIM)     // 65536 reduction dim of the per-step GEMM
#define NKT   (KTOT / 32)      // 2048 k-tiles of 32
#define KSPLIT 16              // split-K across blockIdx.y (16 r values each)

typedef __attribute__((ext_vector_type(16))) __bf16 v16bf;
typedef __attribute__((ext_vector_type(8)))  float  v8f;

union Frag16 {            // one WMMA 16-bit A/B fragment per lane: 16 bf16 = 32B
  v16bf    v;
  uint32_t u[8];
  uint4    q[2];
};

union AccU {              // WMMA f32 C/D fragment: 8 floats per lane
  v8f    v;
  float4 f4[2];
};

__device__ __forceinline__ uint16_t f2bf(float f) {
  uint32_t u = __float_as_uint(f);
  uint32_t r = (u + 0x7FFFu + ((u >> 16) & 1u)) >> 16;  // round-to-nearest-even
  return (uint16_t)r;
}

__device__ __forceinline__ uint32_t pk_mul_bf16(uint32_t a, uint32_t b) {
  uint32_t d;
  asm volatile("v_pk_mul_bf16 %0, %1, %2" : "=v"(d) : "v"(a), "v"(b));
  return d;
}

// WGP-scope prefetch (scope 0): pulls into all cache levels incl. WGP$.
// (__builtin_prefetch only emits SE/SYS scopes, which skip the WGP caches.)
__device__ __forceinline__ void prefetch_wgp(const void* p) {
  asm volatile("global_prefetch_b8 %0, off" :: "v"(p) : "memory");
}

__device__ __forceinline__ v8f wmma_bf16(v16bf a, v16bf b, v8f c) {
  return __builtin_amdgcn_wmma_f32_16x16x32_bf16(
      /*neg_a=*/false, a, /*neg_b=*/false, b,
      /*c_mod=*/(short)0, c, /*reuse_a=*/false, /*reuse_b=*/false);
}

// ---------------------------------------------------------------------------
// pack_T: transition [R][D][R] fp32 -> bf16 WMMA-B fragments.
// Layout: Tpack[((kt*16 + nt)*32 + lane) * 16 .. +15] bf16, where for
// 16-bit B (32x16 KxN, wave32): n = nt*16 + (lane&15); kbase = (lane>=16)?8:0;
// slots 0..7 -> K = kt*32 + kbase + j; slots 8..15 -> K = kt*32 + kbase+16 + j.
// Row index K = r*DIM + d of T.reshape(R*D, R).
// ---------------------------------------------------------------------------
__global__ __launch_bounds__(256) void pack_T_kernel(
    const float* __restrict__ T, uint16_t* __restrict__ Tpack) {
  int idx  = blockIdx.x * 256 + threadIdx.x;   // (kt*16 + nt)*32 + lane
  int lane = idx & 31;
  int nt   = (idx >> 5) & 15;
  int kt   = idx >> 9;
  int n    = nt * 16 + (lane & 15);
  int kb   = (lane >> 4) << 3;
  int k0   = kt * 32 + kb;
  uint16_t vals[16];
#pragma unroll
  for (int j = 0; j < 8; ++j)
    vals[j]     = f2bf(T[(size_t)(k0 + j) * RANK + n]);
#pragma unroll
  for (int j = 0; j < 8; ++j)
    vals[8 + j] = f2bf(T[(size_t)(k0 + 16 + j) * RANK + n]);
  uint4* dst = (uint4*)Tpack + (size_t)idx * 2;
  dst[0] = ((const uint4*)vals)[0];
  dst[1] = ((const uint4*)vals)[1];
}

// ---------------------------------------------------------------------------
// pack_x: x [B][D][L] fp32 -> xT [L][B][D] bf16 via LDS tile transpose.
// Block handles one (b, 64-d chunk): 64 d x 64 i tile.
// ---------------------------------------------------------------------------
__global__ __launch_bounds__(256) void pack_x_kernel(
    const float* __restrict__ x, uint16_t* __restrict__ xT) {
  __shared__ float tile[64 * 65];                 // padded: conflict-free
  int b      = blockIdx.x >> 2;
  int dchunk = (blockIdx.x & 3) * 64;
  int t = threadIdx.x;
  const float* src = x + ((size_t)b * DIM + dchunk) * LEN;
#pragma unroll
  for (int k = 0; k < 16; ++k) {                  // 4096 elems, coalesced read
    int e  = t + k * 256;
    int dl = e >> 6, i = e & 63;
    tile[dl * 65 + i] = src[e];
  }
  __syncthreads();
  int dl = t & 63, iq = t >> 6;
#pragma unroll
  for (int ii = 0; ii < 16; ++ii) {
    int i = iq * 16 + ii;
    xT[((size_t)i * BATCH + b) * DIM + dchunk + dl] = f2bf(tile[dl * 65 + i]);
  }
}

// h0T[n][b] = sum_d x[b,d,0] * alpha[d,n]   (fp32, one-time; transposed state)
__global__ __launch_bounds__(256) void h0_kernel(
    const float* __restrict__ x, const float* __restrict__ alpha,
    float* __restrict__ h0T) {
  int idx = blockIdx.x * 256 + threadIdx.x;
  int b = idx >> 8, n = idx & 255;
  float s = 0.f;
  for (int d = 0; d < DIM; ++d)
    s = fmaf(x[(size_t)b * DIM * LEN + (size_t)d * LEN], alpha[d * RANK + n], s);
  h0T[(size_t)n * BATCH + b] = s;
}

// hTout[idx] = sum over KSPLIT partials (the only writer of next h state)
__global__ __launch_bounds__(256) void reduce_kernel(
    const float* __restrict__ hpart, float* __restrict__ hTout) {
  int idx = blockIdx.x * 256 + threadIdx.x;       // 0 .. R*B-1
  float s = 0.f;
#pragma unroll
  for (int k = 0; k < KSPLIT; ++k)
    s += hpart[(size_t)k * RANK * BATCH + idx];
  hTout[idx] = s;
}

// out[b,o] = sum_r hT[r][b] * omega[r,o]   (fp32, one-time)
__global__ __launch_bounds__(256) void out_kernel(
    const float* __restrict__ hT, const float* __restrict__ omega,
    float* __restrict__ out) {
  int idx = blockIdx.x * 256 + threadIdx.x;
  int b = idx >> 7, o = idx & 127;
  float s = 0.f;
  for (int r = 0; r < RANK; ++r)
    s = fmaf(hT[(size_t)r * BATCH + b], omega[r * OUTD + o], s);
  out[idx] = s;
}

// ---------------------------------------------------------------------------
// One recurrence step, split-K partials (no atomics):
//   hpart[kc][n][b] = partial of KR(h, x_i) @ T over r in [kc*16, kc*16+16).
// Grid: x = 16 output tiles (8 M-blocks of 128 rows x 2 N-blocks of 128 cols),
//       y = KSPLIT. Block: 256 threads = 8 waves (4 wm x 2 wn); each wave owns
// a 32x64 output patch = 2x4 WMMA tiles (64 VGPR f32 accumulators).
// h is kept transposed (hT[R][B]) and pre-converted to packed bf16 pairs in
// LDS once per block, so the inner loop is: 1 ds_load_2addr + 16 v_pk_mul_bf16
// + 8 global_load_b128 + 8 v_wmma + 1 WGP-scope prefetch per iteration.
// ---------------------------------------------------------------------------
__global__ __launch_bounds__(256) void step_kernel(
    const uint16_t* __restrict__ Tpack,
    const uint16_t* __restrict__ xstep,    // [BATCH][DIM] bf16, step i slice
    const float*    __restrict__ hT,       // [RANK][BATCH] fp32
    float*          __restrict__ hpart) {  // [KSPLIT][RANK][BATCH] fp32
  __shared__ uint32_t hlds[16 * 128];      // packed bf16 pair per (rr, row)

  const int tid  = threadIdx.x;
  const int lane = tid & 31;
  const int wave = tid >> 5;
  const int mblk  = blockIdx.x >> 1;       // 0..7  -> rows base mblk*128
  const int nbase = (blockIdx.x & 1) * 128;
  const int r0    = blockIdx.y * 16;
  const int mbase = mblk * 128;

  // Stage h chunk: 16 r x 128 rows, coalesced loads, convert to bf16 pairs once.
#pragma unroll
  for (int c = 0; c < 8; ++c) {
    int idx = tid + c * 256;
    int rr = idx >> 7, row = idx & 127;
    uint32_t hb = f2bf(hT[(size_t)(r0 + rr) * BATCH + mbase + row]);
    hlds[idx] = hb | (hb << 16);
  }
  __syncthreads();

  const int wm = wave >> 1;                // 0..3
  const int wn = wave & 1;                 // 0..1
  const int mrow0 = mbase + wm * 32;       // wave rows [mrow0, mrow0+32)
  const int rloc0 = wm * 32;               // local row base in hlds
  const int ncol0 = nbase + wn * 64;       // wave cols [ncol0, ncol0+64)
  const int ntg0  = ncol0 >> 4;            // first of 4 global n-tiles

  const int mlan = lane & 15;
  const int kb   = (lane >> 4) << 3;       // 0 or 8 (K-slot base per lane half)
  const int b0g  = mrow0 + mlan;
  const int b1g  = mrow0 + 16 + mlan;

  v8f acc[2][4] = {};

  for (int dblk = 0; dblk < 8; ++dblk) {
    const int d0 = dblk * 32;
    // Raw x fragments (bf16), hoisted: reused across all 16 r iterations.
    const uint16_t* xr0 = xstep + (size_t)b0g * DIM + d0 + kb;
    const uint16_t* xr1 = xstep + (size_t)b1g * DIM + d0 + kb;
    const uint4 x0lo = *(const uint4*)(xr0);
    const uint4 x0hi = *(const uint4*)(xr0 + 16);
    const uint4 x1lo = *(const uint4*)(xr1);
    const uint4 x1hi = *(const uint4*)(xr1 + 16);

#pragma unroll 4
    for (int rr = 0; rr < 16; ++rr) {
      const int kt = (r0 + rr) * 8 + dblk; // k-tile index (K = r*DIM + d)

      // Packed bf16(h) pairs from LDS (lanes 0-15 / 16-31 broadcast).
      const uint32_t hb0 = hlds[rr * 128 + rloc0 + mlan];
      const uint32_t hb1 = hlds[rr * 128 + rloc0 + 16 + mlan];

      // A fragments: A[b, r*DIM+d] = h[b,r] * x[b,d], built in registers.
      Frag16 a0, a1;
      a0.u[0] = pk_mul_bf16(x0lo.x, hb0);  a0.u[1] = pk_mul_bf16(x0lo.y, hb0);
      a0.u[2] = pk_mul_bf16(x0lo.z, hb0);  a0.u[3] = pk_mul_bf16(x0lo.w, hb0);
      a0.u[4] = pk_mul_bf16(x0hi.x, hb0);  a0.u[5] = pk_mul_bf16(x0hi.y, hb0);
      a0.u[6] = pk_mul_bf16(x0hi.z, hb0);  a0.u[7] = pk_mul_bf16(x0hi.w, hb0);
      a1.u[0] = pk_mul_bf16(x1lo.x, hb1);  a1.u[1] = pk_mul_bf16(x1lo.y, hb1);
      a1.u[2] = pk_mul_bf16(x1lo.z, hb1);  a1.u[3] = pk_mul_bf16(x1lo.w, hb1);
      a1.u[4] = pk_mul_bf16(x1hi.x, hb1);  a1.u[5] = pk_mul_bf16(x1hi.y, hb1);
      a1.u[6] = pk_mul_bf16(x1hi.z, hb1);  a1.u[7] = pk_mul_bf16(x1hi.w, hb1);

      // B fragments: 4 n-tiles, pre-swizzled, contiguous b128 loads per lane.
      const uint4* bp =
          (const uint4*)Tpack + ((size_t)(kt * 16 + ntg0) * 32 + lane) * 2;
      Frag16 bfr[4];
#pragma unroll
      for (int ni = 0; ni < 4; ++ni) {
        bfr[ni].q[0] = bp[ni * 64];
        bfr[ni].q[1] = bp[ni * 64 + 1];
      }
      // Prefetch next-rr fragments (kt += 8 -> +8192 uint4) into WGP caches.
      prefetch_wgp((const void*)(bp + 8192));

#pragma unroll
      for (int ni = 0; ni < 4; ++ni) {
        acc[0][ni] = wmma_bf16(a0.v, bfr[ni].v, acc[0][ni]);
        acc[1][ni] = wmma_bf16(a1.v, bfr[ni].v, acc[1][ni]);
      }
    }
  }

  // Store partials (transposed layout): lane -> col (lane&15);
  // VGPR v -> row v + 8*(lane>>4)  =>  8 consecutive rows: two b128 stores.
  float* base = hpart + (size_t)blockIdx.y * RANK * BATCH;
  const int rowhi = (lane >> 4) * 8;
#pragma unroll
  for (int mi = 0; mi < 2; ++mi)
#pragma unroll
    for (int ni = 0; ni < 4; ++ni) {
      const int col = ncol0 + ni * 16 + mlan;
      const int row = mrow0 + mi * 16 + rowhi;
      AccU t; t.v = acc[mi][ni];
      float4* dst = (float4*)(base + (size_t)col * BATCH + row);
      dst[0] = t.f4[0];
      dst[1] = t.f4[1];
    }
}

// ---------------------------------------------------------------------------
extern "C" void kernel_launch(void* const* d_in, const int* in_sizes, int n_in,
                              void* d_out, int out_size, void* d_ws, size_t ws_size,
                              hipStream_t stream) {
  const float* x     = (const float*)d_in[0];  // [B, D, L]
  const float* alpha = (const float*)d_in[1];  // [D, R]
  const float* T     = (const float*)d_in[2];  // [R, D, R]
  const float* omega = (const float*)d_in[3];  // [R, O]
  float* out = (float*)d_out;                  // [B, O]

  char* p = (char*)d_ws;
  uint16_t* Tpack = (uint16_t*)p;  p += (size_t)KTOT * RANK * 2;          // 33.5 MB
  uint16_t* xT    = (uint16_t*)p;  p += (size_t)LEN * BATCH * DIM * 2;    // 33.5 MB
  float* hpart    = (float*)p;     p += (size_t)KSPLIT * RANK * BATCH * 4;// 16.8 MB
  float* hbuf0    = (float*)p;     p += (size_t)RANK * BATCH * 4;         // 1 MB
  float* hbuf1    = (float*)p;                                            // 1 MB

  pack_T_kernel<<<(NKT * 16 * 32) / 256, 256, 0, stream>>>(T, Tpack);
  pack_x_kernel<<<BATCH * 4, 256, 0, stream>>>(x, xT);
  h0_kernel<<<(BATCH * RANK) / 256, 256, 0, stream>>>(x, alpha, hbuf0);

  float* hp = hbuf0;
  float* hn = hbuf1;
  for (int i = 1; i < LEN; ++i) {
    step_kernel<<<dim3(16, KSPLIT), 256, 0, stream>>>(
        Tpack, xT + (size_t)i * BATCH * DIM, hp, hpart);
    reduce_kernel<<<(RANK * BATCH) / 256, 256, 0, stream>>>(hpart, hn);
    float* t = hp; hp = hn; hn = t;
  }

  out_kernel<<<(BATCH * OUTD) / 256, 256, 0, stream>>>(hp, omega, out);
}
